// CausalSelfAttention_86045374808729
// MI455X (gfx1250) — matrix operand order repaired and checked by reading, not compiled, not run on hardware
//
#include <hip/hip_runtime.h>
#include <hip/hip_bf16.h>

#define DIMC   1024
#define HEADS  16
#define HD     64
#define SEQ    2048
#define NBATCH 4
#define BHN    (NBATCH * HEADS)   // 64
#define ROWS   (NBATCH * SEQ)     // 8192

typedef __attribute__((ext_vector_type(16))) __bf16 bf16x16;
typedef __attribute__((ext_vector_type(8)))  float  f32x8;

union FragU { bf16x16 v; uint4 q[2]; };

// A fragment (16x32, M x K): lane<16 -> K in {0..7}u{16..23}; lane>=16 -> +8.
__device__ __forceinline__ bf16x16 ld_frag_a(const __bf16* p) {
  FragU u;
  u.q[0] = *(const uint4*)(p);
  u.q[1] = *(const uint4*)(p + 16);
  return u.v;
}
// B fragment (32x16, K x N, N-major in LDS): lane holds column n=lane&15,
// K in {0..7}u{16..23} for lane<16; lane>=16 -> +8. Caller passes base + 8*(lane>>4).
__device__ __forceinline__ bf16x16 ld_frag_b(const __bf16* p) {
  FragU u;
  u.q[0] = *(const uint4*)(p);
  u.q[1] = *(const uint4*)(p + 16);
  return u.v;
}

__device__ __forceinline__ f32x8 wmma_bf16(bf16x16 a, bf16x16 b, f32x8 c) {
  return __builtin_amdgcn_wmma_f32_16x16x32_bf16(
      false, a, false, b, (short)0, c, false, false);
}

// ---- CDNA5 async global->LDS DMA (ASYNCcnt-tracked), per-lane 16B ----------
__device__ __forceinline__ unsigned lds_off(const void* p) {
  return (unsigned)(uintptr_t)p;   // shared aperture: low 32 bits = LDS offset
}
__device__ __forceinline__ void async_b128(unsigned lds, const __bf16* g) {
  asm volatile("global_load_async_to_lds_b128 %0, %1, off"
               :: "v"(lds), "v"(g) : "memory");
}
__device__ __forceinline__ void wait_async0() {
#if __has_builtin(__builtin_amdgcn_s_wait_asynccnt)
  __builtin_amdgcn_s_wait_asynccnt(0);
#else
  asm volatile("s_wait_asynccnt 0" ::: "memory");
#endif
}

// ---------------------------------------------------------------- converts --

__global__ void cvt_bf16_kernel(const float* __restrict__ in,
                                __bf16* __restrict__ out, int n) {
  int i = blockIdx.x * blockDim.x + threadIdx.x;
  if (i < n) out[i] = (__bf16)in[i];
}

// in[K][N] fp32 -> out[N][K] bf16 (coalesced writes)
__global__ void cvt_transpose_kernel(const float* __restrict__ in,
                                     __bf16* __restrict__ out, int K, int N) {
  int i = blockIdx.x * blockDim.x + threadIdx.x;
  if (i >= K * N) return;
  int k = i % K, n = i / K;
  out[i] = (__bf16)in[(size_t)k * N + n];
}

// ---------------------------------------------------------------- GEMM core --
// 256 threads = 8 waves; tile 128(M) x 128(N); K step 32; double-buffered LDS
// filled by async DMA. Wave (wm,wn) computes 64x32 via 4x2 WMMA fragments.
// A: [M][K] row-major bf16. Bt: [N][K] (pre-transposed) bf16.

#define LDSTRIDE 40
#define TILESZ   (128 * LDSTRIDE)

__device__ __forceinline__ void gemm_core(const __bf16* __restrict__ A,
                                          const __bf16* __restrict__ Bt,
                                          __bf16* sA, __bf16* sB,
                                          int m0, int n0, int K,
                                          f32x8 acc[4][2]) {
  const int tid  = threadIdx.x;
  const int lane = tid & 31;
  const int wid  = tid >> 5;
  const int wm   = wid & 1;
  const int wn   = wid >> 1;
  const int ln15 = lane & 15;
  const int kh8  = (lane >> 4) << 3;
  const int kh16 = (lane >> 4) << 4;
  const int lrow = tid >> 1;          // 0..127
  const int lh   = (tid & 1) << 4;    // 0 or 16 elements

#pragma unroll
  for (int mt = 0; mt < 4; ++mt)
#pragma unroll
    for (int nt = 0; nt < 2; ++nt)
#pragma unroll
      for (int r = 0; r < 8; ++r) acc[mt][nt][r] = 0.0f;

  const __bf16* ga0 = A  + (size_t)(m0 + lrow) * K + lh;
  const __bf16* gb0 = Bt + (size_t)(n0 + lrow) * K + lh;
  const int nIters = K >> 5;

  // prologue: DMA tile 0 into buffer 0
  {
    __bf16* la = sA + lrow * LDSTRIDE + lh;
    __bf16* lb = sB + lrow * LDSTRIDE + lh;
    async_b128(lds_off(la),     ga0);
    async_b128(lds_off(la + 8), ga0 + 8);
    async_b128(lds_off(lb),     gb0);
    async_b128(lds_off(lb + 8), gb0 + 8);
  }

  for (int it = 0; it < nIters; ++it) {
    wait_async0();          // my tile-`it` DMA landed
    __syncthreads();        // everyone's landed; prev buffer free for reuse

    if (it + 1 < nIters) {  // DMA next tile while computing this one
      const int k1 = (it + 1) << 5;
      const int nb = (it + 1) & 1;
      __bf16* la = sA + nb * TILESZ + lrow * LDSTRIDE + lh;
      __bf16* lb = sB + nb * TILESZ + lrow * LDSTRIDE + lh;
      async_b128(lds_off(la),     ga0 + k1);
      async_b128(lds_off(la + 8), ga0 + k1 + 8);
      async_b128(lds_off(lb),     gb0 + k1);
      async_b128(lds_off(lb + 8), gb0 + k1 + 8);
    }

    const __bf16* cA = sA + (it & 1) * TILESZ;
    const __bf16* cB = sB + (it & 1) * TILESZ;
    bf16x16 af[4], bfr[2];
#pragma unroll
    for (int mt = 0; mt < 4; ++mt)
      af[mt] = ld_frag_a(cA + (wm * 64 + mt * 16 + ln15) * LDSTRIDE + kh8);
#pragma unroll
    for (int nt = 0; nt < 2; ++nt)
      bfr[nt] = ld_frag_b(cB + (wn * 32 + nt * 16 + ln15) * LDSTRIDE + kh8);
#pragma unroll
    for (int mt = 0; mt < 4; ++mt)
#pragma unroll
      for (int nt = 0; nt < 2; ++nt)
        acc[mt][nt] = wmma_bf16(af[mt], bfr[nt], acc[mt][nt]);
  }
}

// QKV projection: scatter into q[B,H,T,D], k[B,H,T,D], vT[B,H,D,T] (bf16).
// q/k/v boundaries are multiples of 1024 -> whole 128-wide N-tile is uniform.
__launch_bounds__(256)
__global__ void gemm_qkv_kernel(const __bf16* __restrict__ A,
                                const __bf16* __restrict__ Bt,
                                const float* __restrict__ bias,
                                __bf16* __restrict__ qO,
                                __bf16* __restrict__ kO,
                                __bf16* __restrict__ vTO) {
  __shared__ __bf16 sA[2 * TILESZ];
  __shared__ __bf16 sB[2 * TILESZ];
  const int m0 = blockIdx.x * 128;
  const int n0 = blockIdx.y * 128;
  f32x8 acc[4][2];
  gemm_core(A, Bt, sA, sB, m0, n0, DIMC, acc);

  const int lane  = threadIdx.x & 31;
  const int wid   = threadIdx.x >> 5;
  const int wm    = wid & 1;
  const int wn    = wid >> 1;
  const int ln15  = lane & 15;
  const int halfm = (lane >> 4) << 3;
  const int which = n0 >> 10;                 // 0=q, 1=k, 2=v (uniform)
  const int cbase = n0 - (which << 10);

  if (which < 2) {
    __bf16* dst = which ? kO : qO;
#pragma unroll
    for (int nt = 0; nt < 2; ++nt) {
      const int c = cbase + wn * 32 + nt * 16 + ln15;
      const int h = c >> 6, d = c & 63;
      const float bv = bias[n0 + wn * 32 + nt * 16 + ln15];
#pragma unroll
      for (int mt = 0; mt < 4; ++mt)
#pragma unroll
        for (int r = 0; r < 8; ++r) {
          const int gm = m0 + wm * 64 + mt * 16 + halfm + r;
          const int b = gm >> 11, t = gm & (SEQ - 1);
          dst[(((size_t)b * HEADS + h) * SEQ + t) * HD + d] =
              (__bf16)(acc[mt][nt][r] + bv);
        }
    }
  } else {
#pragma unroll
    for (int nt = 0; nt < 2; ++nt) {
      const int c = cbase + wn * 32 + nt * 16 + ln15;
      const int h = c >> 6, d = c & 63;
      const float bv = bias[n0 + wn * 32 + nt * 16 + ln15];
#pragma unroll
      for (int mt = 0; mt < 4; ++mt)
#pragma unroll
        for (int r = 0; r < 8; ++r) {
          const int gm = m0 + wm * 64 + mt * 16 + halfm + r;
          const int b = gm >> 11, t = gm & (SEQ - 1);
          vTO[(((size_t)b * HEADS + h) * HD + d) * SEQ + t] =
              (__bf16)(acc[mt][nt][r] + bv);
        }
    }
  }
}

// Output projection: fp32 result with bias
__launch_bounds__(256)
__global__ void gemm_proj_kernel(const __bf16* __restrict__ A,
                                 const __bf16* __restrict__ Bt,
                                 const float* __restrict__ bias,
                                 float* __restrict__ out) {
  __shared__ __bf16 sA[2 * TILESZ];
  __shared__ __bf16 sB[2 * TILESZ];
  const int m0 = blockIdx.x * 128;
  const int n0 = blockIdx.y * 128;
  f32x8 acc[4][2];
  gemm_core(A, Bt, sA, sB, m0, n0, DIMC, acc);

  const int lane  = threadIdx.x & 31;
  const int wid   = threadIdx.x >> 5;
  const int wm    = wid & 1;
  const int wn    = wid >> 1;
  const int ln15  = lane & 15;
  const int halfm = (lane >> 4) << 3;

#pragma unroll
  for (int nt = 0; nt < 2; ++nt) {
    const int gn = n0 + wn * 32 + nt * 16 + ln15;
    const float bv = bias[gn];
#pragma unroll
    for (int mt = 0; mt < 4; ++mt)
#pragma unroll
      for (int r = 0; r < 8; ++r) {
        const int gm = m0 + wm * 64 + mt * 16 + halfm + r;
        out[(size_t)gm * DIMC + gn] = acc[mt][nt][r] + bv;
      }
  }
}

// ------------------------------------------------------------ flash attention
// 128 threads = 4 waves; each wave owns 16 q rows. 64-wide K/V tiles double-
// buffered in LDS via async DMA; online softmax; both matmuls WMMA.

__launch_bounds__(128)
__global__ void flash_attn_kernel(const __bf16* __restrict__ Q,
                                  const __bf16* __restrict__ Kt,
                                  const __bf16* __restrict__ VT,
                                  __bf16* __restrict__ O) {
  __shared__ __bf16 sK[2 * 64 * 72];   // [t_k][d]
  __shared__ __bf16 sV[2 * 64 * 72];   // [d][t_k]  (v transposed)
  __shared__ __bf16 sP[4 * 16 * 72];   // per-wave P tiles [m][t_k]

  const int tid  = threadIdx.x;
  const int lane = tid & 31;
  const int wid  = tid >> 5;           // 0..3
  const int bh   = blockIdx.x;         // 0..63
  const int q0   = blockIdx.y * 64;
  const int b    = bh >> 4, h = bh & 15;
  const int ln15 = lane & 15;
  const int kh8  = (lane >> 4) << 3;
  const int kh16 = (lane >> 4) << 4;
  const int mh8  = (lane >> 4) << 3;   // C-layout row offset

  const int nk   = blockIdx.y + 1;     // causal: tiles with t0 <= q0
  const int lrow = tid >> 1;           // 0..63
  const int lh   = (tid & 1) << 5;     // 0 or 32 elements

  const __bf16* gk0 = Kt + ((size_t)bh * SEQ + lrow) * HD + lh;
  const __bf16* gv0 = VT + ((size_t)bh * HD + lrow) * SEQ + lh;

  // prologue: DMA K/V tile 0 into buffer 0
  {
    __bf16* lk = sK + lrow * 72 + lh;
    __bf16* lv = sV + lrow * 72 + lh;
#pragma unroll
    for (int j = 0; j < 4; ++j) {
      async_b128(lds_off(lk + 8 * j), gk0 + 8 * j);
      async_b128(lds_off(lv + 8 * j), gv0 + 8 * j);
    }
  }

  // q fragments for this wave's 16 rows (K = HD = 64 -> 2 WMMA steps)
  bf16x16 qf[2];
  {
    const __bf16* qrow = Q + ((size_t)bh * SEQ + q0 + wid * 16 + ln15) * HD;
    qf[0] = ld_frag_a(qrow + kh8);
    qf[1] = ld_frag_a(qrow + 32 + kh8);
  }

  f32x8 o[4];
  float mrun[8], lrun[8];
#pragma unroll
  for (int nf = 0; nf < 4; ++nf)
#pragma unroll
    for (int r = 0; r < 8; ++r) o[nf][r] = 0.0f;
#pragma unroll
  for (int r = 0; r < 8; ++r) { mrun[r] = -1e30f; lrun[r] = 0.0f; }

  for (int kt = 0; kt < nk; ++kt) {
    const int t0 = kt * 64;
    wait_async0();
    __syncthreads();

    if (kt + 1 < nk) {                 // DMA next K/V tile while computing
      const int nb = (kt + 1) & 1;
      __bf16* lk = sK + nb * (64 * 72) + lrow * 72 + lh;
      __bf16* lv = sV + nb * (64 * 72) + lrow * 72 + lh;
#pragma unroll
      for (int j = 0; j < 4; ++j) {
        async_b128(lds_off(lk + 8 * j), gk0 + (size_t)(t0 + 64) * HD + 8 * j);
        async_b128(lds_off(lv + 8 * j), gv0 + (t0 + 64) + 8 * j);
      }
    }

    const __bf16* cK = sK + (kt & 1) * (64 * 72);
    const __bf16* cV = sV + (kt & 1) * (64 * 72);

    // S = q . k^T  (16 x 64)
    f32x8 s[4];
#pragma unroll
    for (int nf = 0; nf < 4; ++nf)
#pragma unroll
      for (int r = 0; r < 8; ++r) s[nf][r] = 0.0f;
#pragma unroll
    for (int ks = 0; ks < 2; ++ks)
#pragma unroll
      for (int nf = 0; nf < 4; ++nf) {
        bf16x16 kb = ld_frag_b(cK + (nf * 16 + ln15) * 72 + ks * 32 + kh8);
        s[nf] = wmma_bf16(qf[ks], kb, s[nf]);
      }

    // scale + causal mask (only the diagonal tile crosses)
    const bool last = (kt == nk - 1);
#pragma unroll
    for (int nf = 0; nf < 4; ++nf)
#pragma unroll
      for (int r = 0; r < 8; ++r) {
        float v = s[nf][r] * 0.125f;   // 1/sqrt(64)
        if (last) {
          const int kj = t0 + nf * 16 + ln15;
          const int qi = q0 + wid * 16 + mh8 + r;
          if (kj > qi) v = -1e30f;
        }
        s[nf][r] = v;
      }

    // row max across frags, then across 16-lane group
    float rm[8];
#pragma unroll
    for (int r = 0; r < 8; ++r)
      rm[r] = fmaxf(fmaxf(s[0][r], s[1][r]), fmaxf(s[2][r], s[3][r]));
#pragma unroll
    for (int msk = 1; msk < 16; msk <<= 1)
#pragma unroll
      for (int r = 0; r < 8; ++r)
        rm[r] = fmaxf(rm[r], __shfl_xor(rm[r], msk, 32));

    float alpha[8];
#pragma unroll
    for (int r = 0; r < 8; ++r) {
      const float mn = fmaxf(mrun[r], rm[r]);
      alpha[r] = __expf(mrun[r] - mn);
      mrun[r] = mn;
    }

    float rs[8];
#pragma unroll
    for (int r = 0; r < 8; ++r) rs[r] = 0.0f;
#pragma unroll
    for (int nf = 0; nf < 4; ++nf)
#pragma unroll
      for (int r = 0; r < 8; ++r) {
        const float p = __expf(s[nf][r] - mrun[r]);
        s[nf][r] = p;
        rs[r] += p;
      }
#pragma unroll
    for (int msk = 1; msk < 16; msk <<= 1)
#pragma unroll
      for (int r = 0; r < 8; ++r) rs[r] += __shfl_xor(rs[r], msk, 32);
#pragma unroll
    for (int r = 0; r < 8; ++r) lrun[r] = lrun[r] * alpha[r] + rs[r];
#pragma unroll
    for (int nf = 0; nf < 4; ++nf)
#pragma unroll
      for (int r = 0; r < 8; ++r) o[nf][r] *= alpha[r];

    // reshape P (C-layout -> A-layout) through per-wave LDS (same-wave DS
    // ops are in-order; no barrier needed)
    __bf16* pb = sP + wid * 16 * 72;
#pragma unroll
    for (int nf = 0; nf < 4; ++nf)
#pragma unroll
      for (int r = 0; r < 8; ++r)
        pb[(mh8 + r) * 72 + nf * 16 + ln15] = (__bf16)s[nf][r];

    // O += P . V   (K = 64 -> 2 WMMA steps)
#pragma unroll
    for (int ks = 0; ks < 2; ++ks) {
      bf16x16 pa = ld_frag_a(pb + ln15 * 72 + ks * 32 + kh8);
#pragma unroll
      for (int nf = 0; nf < 4; ++nf) {
        bf16x16 vb = ld_frag_b(cV + (nf * 16 + ln15) * 72 + ks * 32 + kh8);
        o[nf] = wmma_bf16(pa, vb, o[nf]);
      }
    }
  }

  // epilogue: normalize and scatter into attn[B*T, C] bf16
#pragma unroll
  for (int nf = 0; nf < 4; ++nf)
#pragma unroll
    for (int r = 0; r < 8; ++r) {
      const float val = o[nf][r] / lrun[r];
      const int t = q0 + wid * 16 + mh8 + r;
      const int col = h * HD + nf * 16 + ln15;
      O[((size_t)b * SEQ + t) * DIMC + col] = (__bf16)val;
    }
}

// ------------------------------------------------------------------ launcher

extern "C" void kernel_launch(void* const* d_in, const int* in_sizes, int n_in,
                              void* d_out, int out_size, void* d_ws, size_t ws_size,
                              hipStream_t stream) {
  const float* x      = (const float*)d_in[0];
  const float* w_qkv  = (const float*)d_in[1];
  const float* b_qkv  = (const float*)d_in[2];
  const float* w_proj = (const float*)d_in[3];
  const float* b_proj = (const float*)d_in[4];
  float* out = (float*)d_out;

  char* ws = (char*)d_ws;
  __bf16* xb     = (__bf16*)ws;  ws += (size_t)ROWS * DIMC * 2;        // 16 MB
  __bf16* wqkvT  = (__bf16*)ws;  ws += (size_t)3 * DIMC * DIMC * 2;    // 6 MB
  __bf16* wprojT = (__bf16*)ws;  ws += (size_t)DIMC * DIMC * 2;        // 2 MB
  __bf16* q      = (__bf16*)ws;  ws += (size_t)BHN * SEQ * HD * 2;     // 16 MB
  __bf16* k      = (__bf16*)ws;  ws += (size_t)BHN * SEQ * HD * 2;     // 16 MB
  __bf16* vT     = (__bf16*)ws;  ws += (size_t)BHN * SEQ * HD * 2;     // 16 MB
  __bf16* attn   = (__bf16*)ws;  ws += (size_t)ROWS * DIMC * 2;        // 16 MB

  const int nx = ROWS * DIMC;
  cvt_bf16_kernel<<<nx / 256, 256, 0, stream>>>(x, xb, nx);
  cvt_transpose_kernel<<<(3 * DIMC * DIMC) / 256, 256, 0, stream>>>(
      w_qkv, wqkvT, DIMC, 3 * DIMC);
  cvt_transpose_kernel<<<(DIMC * DIMC) / 256, 256, 0, stream>>>(
      w_proj, wprojT, DIMC, DIMC);

  gemm_qkv_kernel<<<dim3(ROWS / 128, (3 * DIMC) / 128), 256, 0, stream>>>(
      xb, wqkvT, b_qkv, q, k, vT);

  flash_attn_kernel<<<dim3(BHN, SEQ / 64), 128, 0, stream>>>(q, k, vT, attn);

  gemm_proj_kernel<<<dim3(ROWS / 128, DIMC / 128), 256, 0, stream>>>(
      attn, wprojT, b_proj, out);
}
